// BahdanauAttention_65068754534740
// MI455X (gfx1250) — matrix operand order repaired and checked by reading, not compiled, not run on hardware
//
#include <hip/hip_runtime.h>
#include <hip/hip_bf16.h>
#include <cstdint>

// ---------------------------------------------------------------------------
// Bahdanau attention for MI455X (gfx1250, wave32).
//   B=64, T=2048, D=512, U=512
// Dominant cost: features@W1 GEMM (68.7 GFLOP) fused with tanh/V-dot so the
// [B,T,U] tensor is never written. fp32 -> bf16 on the fly, accumulate fp32
// with v_wmma_f32_16x16x32_bf16. Register blocking 4m x 2u per wave:
//   - each B fragment feeds 4 WMMAs (0.5 global b128 / wmma)
//   - each A fragment feeds 2 WMMAs (1 ds b128 / wmma)
// ---------------------------------------------------------------------------

#define BB 64
#define TT 2048
#define DD 512
#define UU 512

typedef __attribute__((ext_vector_type(16))) __bf16        v16bf;
typedef __attribute__((ext_vector_type(8)))  float         v8f;
typedef __attribute__((ext_vector_type(8)))  unsigned int  v8u;

// round-to-nearest-even fp32 -> bf16, pack two into one dword (lo = first)
__device__ __forceinline__ unsigned int pack_bf16(float lo, float hi) {
  unsigned int xl = __builtin_bit_cast(unsigned int, lo);
  unsigned int xh = __builtin_bit_cast(unsigned int, hi);
  xl = xl + 0x7fffu + ((xl >> 16) & 1u);
  xh = xh + 0x7fffu + ((xh >> 16) & 1u);
  return (xl >> 16) | (xh & 0xffff0000u);
}

// ---------------------------------------------------------------------------
// Repack W1 [D=512 rows (k) x U=512 cols] fp32 -> bf16 B-fragment-major.
// B-matrix (32x16, 16-bit) per-wave layout: n = lane & 15,
// dword v (0..7) packs K = 8*(lane>>4) + (v>=4?16:0) + (v&3)*2 (lo) and +1 (hi).
// Storage: frag[u_tile(32)][k_step(16)][lane(32)][v(8)]  = 512 KB total.
// ---------------------------------------------------------------------------
__global__ __launch_bounds__(256) void k_prep_w1(const float* __restrict__ W1,
                                                 unsigned int* __restrict__ w1f) {
  int tid  = blockIdx.x * 256 + threadIdx.x;   // 0 .. 131071
  int v    = tid & 7;
  int lane = (tid >> 3) & 31;
  int ks   = (tid >> 8) & 15;
  int ut   = tid >> 12;
  int k = ks * 32 + ((lane >> 4) << 3) + ((v & 4) << 2) + ((v & 3) << 1);
  int u = ut * 16 + (lane & 15);
  const float* p = W1 + (size_t)k * UU + u;
  w1f[tid] = pack_bf16(p[0], p[UU]);
}

// ---------------------------------------------------------------------------
// hcomb[b][u] = hidden[b] @ W2[:,u] + W2_b[u] + W1_b[u]   (tiny GEMM)
// ---------------------------------------------------------------------------
__global__ __launch_bounds__(256) void k_hproj(const float* __restrict__ hidden,
                                               const float* __restrict__ W2,
                                               const float* __restrict__ W2b,
                                               const float* __restrict__ W1b,
                                               float* __restrict__ hcomb) {
  int idx = blockIdx.x * 256 + threadIdx.x;    // 0 .. 32767
  int b = idx >> 9;
  int u = idx & (UU - 1);
  float acc = W2b[u] + W1b[u];
  const float* h = hidden + (size_t)b * DD;
#pragma unroll 8
  for (int k = 0; k < DD; ++k) acc += h[k] * W2[(size_t)k * UU + u];
  hcomb[idx] = acc;
}

// ---------------------------------------------------------------------------
// score[b][t] = V . tanh(features[b][t] @ W1 + hcomb[b])  (fused, WMMA)
// One block = one (b, 64-row t-tile): 4 m-tiles x 32 u-tiles.
// 8 waves; wave w owns u-tiles {4w .. 4w+3}, processed as 2 pairs.
// Per pair: 8 accumulators; each k-step does 4 A loads + 2 B loads + 8 WMMA.
// 256 WMMA per wave, 2048 per block.
// ---------------------------------------------------------------------------
__global__ __launch_bounds__(256) void k_scores(const float* __restrict__ feat,
                                                const unsigned int* __restrict__ w1f,
                                                const float* __restrict__ hcomb,
                                                const float* __restrict__ Vw,
                                                const float* __restrict__ Vb,
                                                float* __restrict__ scores) {
  __shared__ unsigned int sA[64 * 256];   // 64 t-rows x 512 bf16 (packed) = 64 KB
  __shared__ float        sH[UU];         // hcomb[b]
  __shared__ float        sScore[64];

  int tid = threadIdx.x;
  int b   = blockIdx.x >> 5;              // 32 t-tiles of 64 per batch row
  int t0  = (blockIdx.x & 31) << 6;

  // Stage features tile -> bf16 in LDS (coalesced float2 loads).
  const float* fbase = feat + ((size_t)b * TT + t0) * DD;
#pragma unroll 4
  for (int i = 0; i < 64; ++i) {
    int idx = i * 256 + tid;              // consecutive threads -> consecutive dwords
    int row = idx >> 8;
    int cd  = idx & 255;
    float2 f = *(const float2*)(fbase + (size_t)row * DD + 2 * cd);
    sA[idx] = pack_bf16(f.x, f.y);
  }
  sH[tid]       = hcomb[(size_t)b * UU + tid];
  sH[tid + 256] = hcomb[(size_t)b * UU + tid + 256];
  if (tid < 64) sScore[tid] = 0.0f;
  __syncthreads();

  int wave  = tid >> 5;
  int lane  = tid & 31;
  int n     = lane & 15;                  // C/D: column within u-tile
  int hi    = lane >> 4;                  // C/D: 0 -> rows 0..7, 1 -> rows 8..15
  int m     = n;                          // A frag: row-in-mtile = lane & 15
  int khalf = hi ? 8 : 0;                 // A frag: lanes 16..31 hold K+8

  float partial[4][8];
#pragma unroll
  for (int mi = 0; mi < 4; ++mi)
#pragma unroll
    for (int r = 0; r < 8; ++r) partial[mi][r] = 0.0f;

  for (int uts = 0; uts < 2; ++uts) {
    int ut0 = wave * 4 + uts * 2;         // u-tile pair (ut0, ut0+1)
    v8f acc[2][4];
#pragma unroll
    for (int q = 0; q < 2; ++q)
#pragma unroll
      for (int mi = 0; mi < 4; ++mi) acc[q][mi] = {};

    const unsigned int* bBase0 = w1f + (((size_t)ut0 * 16) * 32 + lane) * 8;
    const unsigned int* bBase1 = bBase0 + (size_t)16 * 32 * 8;

    for (int ks = 0; ks < 16; ++ks) {
      const unsigned int* bp0 = bBase0 + (size_t)ks * 256;
      const unsigned int* bp1 = bBase1 + (size_t)ks * 256;
      if (ks < 15) {
        __builtin_prefetch(bp0 + 256, 0, 1);   // global_prefetch next B frags
        __builtin_prefetch(bp1 + 256, 0, 1);
      }
      v8u bu0 = *(const v8u*)bp0;              // 2x b128 global loads each
      v8u bu1 = *(const v8u*)bp1;
      int kb = ks * 32;

      // Each A fragment amortized over 2 u-tiles; each B over 4 m-tiles.
#pragma unroll
      for (int mi = 0; mi < 4; ++mi) {
        // A fragment from LDS per 16-bit 16x32 layout:
        //   dword v of lane holds K = khalf + (v>=4?16:0) + (v&3)*2 (+1 hi)
        v8u au;
#pragma unroll
        for (int v = 0; v < 8; ++v) {
          int koff = khalf + ((v & 4) << 2) + ((v & 3) << 1);
          au[v] = sA[(mi * 16 + m) * 256 + ((kb + koff) >> 1)];
        }
        acc[0][mi] = __builtin_amdgcn_wmma_f32_16x16x32_bf16(
            false, __builtin_bit_cast(v16bf, au),
            false, __builtin_bit_cast(v16bf, bu0),
            (short)0, acc[0][mi], false, false);
        acc[1][mi] = __builtin_amdgcn_wmma_f32_16x16x32_bf16(
            false, __builtin_bit_cast(v16bf, au),
            false, __builtin_bit_cast(v16bf, bu1),
            (short)0, acc[1][mi], false, false);
      }
    }

    // epilogue for this u-tile pair: tanh(f_proj + hcomb) * V, summed over u
#pragma unroll
    for (int q = 0; q < 2; ++q) {
      int u = (ut0 + q) * 16 + n;
      float hv = sH[u];
      float vw = Vw[u];
#pragma unroll
      for (int mi = 0; mi < 4; ++mi)
#pragma unroll
        for (int r = 0; r < 8; ++r)
          partial[mi][r] += tanhf(acc[q][mi][r] + hv) * vw;
    }
  }

  // Reduce over the 16 n-lanes of each half-wave, then over waves via LDS.
#pragma unroll
  for (int mi = 0; mi < 4; ++mi)
#pragma unroll
    for (int r = 0; r < 8; ++r) {
      float p = partial[mi][r];
      p += __shfl_xor(p, 1, 32);
      p += __shfl_xor(p, 2, 32);
      p += __shfl_xor(p, 4, 32);
      p += __shfl_xor(p, 8, 32);
      if (n == 0) atomicAdd(&sScore[mi * 16 + r + 8 * hi], p);   // ds_add_f32
    }
  __syncthreads();
  if (tid < 64) scores[(size_t)b * TT + t0 + tid] = sScore[tid] + Vb[0];
}

// ---------------------------------------------------------------------------
// Softmax over T per batch row (one block per b).
// ---------------------------------------------------------------------------
__global__ __launch_bounds__(256) void k_softmax(const float* __restrict__ scores,
                                                 float* __restrict__ attn) {
  __shared__ float red[256];
  int b = blockIdx.x, tid = threadIdx.x;
  const float* s = scores + (size_t)b * TT;
  float e[8];
  float mx = -3.4e38f;
#pragma unroll
  for (int i = 0; i < 8; ++i) { e[i] = s[tid + i * 256]; mx = fmaxf(mx, e[i]); }
  red[tid] = mx; __syncthreads();
  for (int off = 128; off > 0; off >>= 1) {
    if (tid < off) red[tid] = fmaxf(red[tid], red[tid + off]);
    __syncthreads();
  }
  float gm = red[0]; __syncthreads();
  float sum = 0.f;
#pragma unroll
  for (int i = 0; i < 8; ++i) { e[i] = __expf(e[i] - gm); sum += e[i]; }
  red[tid] = sum; __syncthreads();
  for (int off = 128; off > 0; off >>= 1) {
    if (tid < off) red[tid] += red[tid + off];
    __syncthreads();
  }
  float inv = 1.0f / red[0];
  float* a = attn + (size_t)b * TT;
#pragma unroll
  for (int i = 0; i < 8; ++i) a[tid + i * 256] = e[i] * inv;
}

__global__ __launch_bounds__(256) void k_zero(float* __restrict__ p, int n) {
  int i = blockIdx.x * 256 + threadIdx.x;
  if (i < n) p[i] = 0.0f;
}

// ---------------------------------------------------------------------------
// context[b][d] = sum_t attn[b][t] * features[b][t][d]
// 16 blocks per batch row (t split) -> 1024 blocks to saturate 23.3 TB/s HBM.
// ---------------------------------------------------------------------------
__global__ __launch_bounds__(256) void k_context(const float* __restrict__ feat,
                                                 const float* __restrict__ attn,
                                                 float* __restrict__ out) {
  __shared__ float sAt[128];
  int b   = blockIdx.x >> 4;
  int ts  = (blockIdx.x & 15) * 128;
  int tid = threadIdx.x;
  if (tid < 128) sAt[tid] = attn[(size_t)b * TT + ts + tid];
  __syncthreads();
  const float* fb = feat + ((size_t)b * TT + ts) * DD;
  float a0 = 0.f, a1 = 0.f;
  for (int t = 0; t < 128; ++t) {
    float w = sAt[t];
    a0 += w * fb[(size_t)t * DD + tid];         // coalesced across threads
    a1 += w * fb[(size_t)t * DD + tid + 256];
  }
  atomicAdd(&out[b * DD + tid], a0);
  atomicAdd(&out[b * DD + tid + 256], a1);
}

// ---------------------------------------------------------------------------
extern "C" void kernel_launch(void* const* d_in, const int* in_sizes, int n_in,
                              void* d_out, int out_size, void* d_ws, size_t ws_size,
                              hipStream_t stream) {
  (void)in_sizes; (void)n_in; (void)out_size; (void)ws_size;
  const float* features = (const float*)d_in[0];
  const float* hidden   = (const float*)d_in[1];
  const float* W1w      = (const float*)d_in[2];
  const float* W1b      = (const float*)d_in[3];
  const float* W2w      = (const float*)d_in[4];
  const float* W2b      = (const float*)d_in[5];
  const float* Vw       = (const float*)d_in[6];
  const float* Vb       = (const float*)d_in[7];
  float* out = (float*)d_out;

  char* ws = (char*)d_ws;
  unsigned int* w1f    = (unsigned int*)ws;               // 512 KB bf16 W1 fragments
  float*        hcomb  = (float*)(ws + (512u << 10));     // 128 KB
  float*        scores = (float*)(ws + (640u << 10));     // 512 KB
  float*        attn   = (float*)(ws + (1152u << 10));    // 512 KB

  k_prep_w1<<<512, 256, 0, stream>>>(W1w, w1f);
  k_hproj  <<<128, 256, 0, stream>>>(hidden, W2w, W2b, W1b, hcomb);
  k_scores <<<BB * (TT / 64), 256, 0, stream>>>(features, w1f, hcomb, Vw, Vb, scores);
  k_softmax<<<BB, 256, 0, stream>>>(scores, attn);
  k_zero   <<<(BB * DD + 255) / 256, 256, 0, stream>>>(out, BB * DD);
  k_context<<<BB * 16, 256, 0, stream>>>(features, attn, out);
}
